// AdaptiveQLSTM_65481071395287
// MI455X (gfx1250) — compile-verified, matrix operand order checked by reading, not run on hardware
//
#include <hip/hip_runtime.h>
#include <hip/hip_bf16.h>

// Problem dims (fixed by the reference)
#define T_STEPS 1024
#define BATCH   64
#define DIN     512
#define HID     512
#define KDIM    1024   // D + H
#define NDIM    2048   // 4 * H

typedef __attribute__((ext_vector_type(16))) __bf16 v16bf;
typedef __attribute__((ext_vector_type(8)))  float  v8f;
typedef __attribute__((ext_vector_type(4)))  unsigned int v4u;
typedef __attribute__((ext_vector_type(8)))  int v8i;
typedef __attribute__((ext_vector_type(4)))  int v4i;

struct Frag32B { uint4 lo, hi; };   // 32 bytes = 16 bf16 values

// Load a 16-half WMMA fragment slice for this lane.
// Layout per CDNA5 ISA 7.12.2 (16-bit 16x32 operands): this lane holds halves
// [kbase .. kbase+7] and [kbase+16 .. kbase+23] of its row (A) / column (B).
__device__ __forceinline__ v16bf load_frag(const __hip_bfloat16* p) {
    Frag32B f;
    f.lo = *reinterpret_cast<const uint4*>(p);        // K = kbase .. kbase+7
    f.hi = *reinterpret_cast<const uint4*>(p + 16);   // K = kbase+16 .. kbase+23
    return __builtin_bit_cast(v16bf, f);
}

__device__ __forceinline__ float fast_sigmoid(float v) {
    return 1.0f / (1.0f + __expf(-v));
}
__device__ __forceinline__ float fast_tanh(float v) {
    float e = __expf(2.0f * v);
    return (e - 1.0f) / (e + 1.0f);
}

// ---------------------------------------------------------------------------
// Prep: transposed bf16 fused weight Wt[n][k] = W[k][n] and fused bias.
// ---------------------------------------------------------------------------
__global__ void lstm_pack_weights(const float* __restrict__ Wf, const float* __restrict__ Wi,
                                  const float* __restrict__ Wg, const float* __restrict__ Wo,
                                  const float* __restrict__ bf_, const float* __restrict__ bi_,
                                  const float* __restrict__ bg_, const float* __restrict__ bo_,
                                  __hip_bfloat16* __restrict__ Wt, float* __restrict__ bias) {
    int idx = blockIdx.x * blockDim.x + threadIdx.x;      // 0 .. 2048*1024-1
    if (idx >= NDIM * KDIM) return;
    int n = idx >> 10;          // 0..2047
    int k = idx & (KDIM - 1);   // 0..1023
    int gate = n >> 9;
    int col  = n & (HID - 1);
    const float* Wsrc = (gate == 0) ? Wf : (gate == 1) ? Wi : (gate == 2) ? Wg : Wo;
    Wt[(size_t)n * KDIM + k] = __float2bfloat16(Wsrc[(size_t)k * HID + col]);
    if (k == 0) {
        const float* bsrc = (gate == 0) ? bf_ : (gate == 1) ? bi_ : (gate == 2) ? bg_ : bo_;
        bias[n] = bsrc[col];
    }
}

// ---------------------------------------------------------------------------
// Prep: xh0[b][0:512] = bf16(x[0,b,:]), xh0[b][512:1024] = 0 (h0), c0 = 0.
// ---------------------------------------------------------------------------
__global__ void lstm_init_state(const float* __restrict__ x_all,
                                __hip_bfloat16* __restrict__ xh,
                                float* __restrict__ c) {
    int idx = blockIdx.x * blockDim.x + threadIdx.x;      // 0 .. 64*512-1
    if (idx >= BATCH * HID) return;
    int b = idx >> 9, col = idx & (HID - 1);
    xh[(size_t)b * KDIM + col]        = __float2bfloat16(x_all[idx]);  // t = 0 slice
    xh[(size_t)b * KDIM + HID + col]  = __float2bfloat16(0.0f);
    c[idx] = 0.0f;
}

// ---------------------------------------------------------------------------
// TDM helper: issue tensor_load_to_lds of a 2D tile (ISA 7.7 / D# 8.3-8.6).
//  - 16 rows x 512 dwords (= 16 x 1024 bf16), row stride 512 dwords,
//    source tensor 64 x 512 dwords (the whole xh state).
//  - groups 2/3 zero (<=2D tensor). 6-arg builtin form for this toolchain.
// ---------------------------------------------------------------------------
__device__ __forceinline__ void tdm_load_tile_2d(unsigned lds_addr,
                                                 const void* gaddr) {
    unsigned long long ga = (unsigned long long)(uintptr_t)gaddr;

    v4u g0;
    g0[0] = 1u;                                            // count=1, user mode
    g0[1] = lds_addr;                                      // lds_addr (bytes)
    g0[2] = (unsigned)(ga & 0xFFFFFFFFull);                // global_addr[31:0]
    g0[3] = (unsigned)((ga >> 32) & 0x01FFFFFFull)         // global_addr[56:32]
          | (2u << 30);                                    // type = 2 ("image")

    const unsigned td0 = 512, td1 = 64;                    // tensor dims (dword units)
    const unsigned tl0 = 512, tl1 = 16, tl2 = 0;           // tile dims
    const unsigned long long s0 = 512;                     // dim0 stride (dwords)
    v8i g1;
    g1[0] = (int)(2u << 16);                               // data_size = 2 (4 bytes)
    g1[1] = (int)((td0 & 0xFFFFu) << 16);                  // [63:48] = tensor_dim0 lo
    g1[2] = (int)((td0 >> 16) | ((td1 & 0xFFFFu) << 16));  // td0 hi | td1 lo
    g1[3] = (int)((td1 >> 16) | (tl0 << 16));              // td1 hi | tile_dim0
    g1[4] = (int)(tl1 | (tl2 << 16));                      // tile_dim1 | tile_dim2
    g1[5] = (int)(s0 & 0xFFFFFFFFull);                     // dim0_stride[31:0]
    g1[6] = (int)(s0 >> 32);                               // stride hi | stride1 lo
    g1[7] = 0;

    v4i z4 = {0, 0, 0, 0};
    v8i z8 = {0, 0, 0, 0, 0, 0, 0, 0};
    __builtin_amdgcn_tensor_load_to_lds(g0, g1, z4, z4, z8, 0);
}

// ---------------------------------------------------------------------------
// Fused LSTM step. One wave owns a 16(batch) x 16(hidden) tile; its four WMMA
// accumulators are the f/i/g/o pre-activations for the SAME elements, so
// gates + cell update happen in-register. The block's shared A tile
// (16 rows x 1024 bf16 of xh) is DMA'd into LDS once by the Tensor Data
// Mover; fragments then come from ds_load_b128. B (Wt) streams from L2.
// Reads xhA/cA (state t), writes xhB/cB (state t+1): ping-pong, race-free.
// ---------------------------------------------------------------------------
__global__ void __launch_bounds__(256)
lstm_step(const __hip_bfloat16* __restrict__ xhA,
          __hip_bfloat16* __restrict__ xhB,
          const __hip_bfloat16* __restrict__ Wt,
          const float* __restrict__ bias,
          const float* __restrict__ cA,
          float* __restrict__ cB,
          const float* __restrict__ x_all,
          float* __restrict__ out,
          int t,
          float* __restrict__ hx_out, float* __restrict__ cx_out) {
    __shared__ __hip_bfloat16 smA[16 * KDIM];              // 32 KiB block A tile

    const int lane  = threadIdx.x & 31;
    const int wid   = threadIdx.x >> 5;                    // wave in block 0..7
    const int m0    = (blockIdx.x >> 2) * 16;              // batch tile (uniform/block)
    const int col0  = ((blockIdx.x & 3) * 8 + wid) * 16;   // hidden tile per wave
    const int row   = lane & 15;
    const int khalf = (lane >> 4) * 8;                     // lanes 16-31: K+8 halves

    // --- TDM: stage A tile (rows m0..m0+15 of xhA) into LDS ---------------
    if (wid == 0) {
        tdm_load_tile_2d((unsigned)(uintptr_t)(void*)smA,
                         xhA + (size_t)m0 * KDIM);
        __builtin_amdgcn_s_wait_tensorcnt(0);
    }
    __syncthreads();

    const __hip_bfloat16* arow = smA + (size_t)row * KDIM;           // LDS
    const __hip_bfloat16* wrow = Wt + (size_t)(col0 + row) * KDIM;   // global (L2)
    const int G1 = 512 * KDIM, G2 = 1024 * KDIM, G3 = 1536 * KDIM;   // gate offsets

    v8f acc[4] = {v8f{}, v8f{}, v8f{}, v8f{}};

    // --- software pipeline: stage fragments for kk = 0 ---
    v16bf a_c  = load_frag(arow + khalf);
    v16bf w_c0 = load_frag(wrow + khalf);
    v16bf w_c1 = load_frag(wrow + G1 + khalf);
    v16bf w_c2 = load_frag(wrow + G2 + khalf);
    v16bf w_c3 = load_frag(wrow + G3 + khalf);

    #pragma unroll 2
    for (int kk = 32; kk < KDIM; kk += 32) {
        const int kb = kk + khalf;
        // issue next-iteration loads BEFORE consuming current fragments
        v16bf a_n  = load_frag(arow + kb);
        v16bf w_n0 = load_frag(wrow + kb);
        v16bf w_n1 = load_frag(wrow + G1 + kb);
        v16bf w_n2 = load_frag(wrow + G2 + kb);
        v16bf w_n3 = load_frag(wrow + G3 + kb);
        acc[0] = __builtin_amdgcn_wmma_f32_16x16x32_bf16(false, a_c, false, w_c0, (short)0, acc[0], false, false);
        acc[1] = __builtin_amdgcn_wmma_f32_16x16x32_bf16(false, a_c, false, w_c1, (short)0, acc[1], false, false);
        acc[2] = __builtin_amdgcn_wmma_f32_16x16x32_bf16(false, a_c, false, w_c2, (short)0, acc[2], false, false);
        acc[3] = __builtin_amdgcn_wmma_f32_16x16x32_bf16(false, a_c, false, w_c3, (short)0, acc[3], false, false);
        a_c = a_n; w_c0 = w_n0; w_c1 = w_n1; w_c2 = w_n2; w_c3 = w_n3;
    }
    // epilogue k-step
    acc[0] = __builtin_amdgcn_wmma_f32_16x16x32_bf16(false, a_c, false, w_c0, (short)0, acc[0], false, false);
    acc[1] = __builtin_amdgcn_wmma_f32_16x16x32_bf16(false, a_c, false, w_c1, (short)0, acc[1], false, false);
    acc[2] = __builtin_amdgcn_wmma_f32_16x16x32_bf16(false, a_c, false, w_c2, (short)0, acc[2], false, false);
    acc[3] = __builtin_amdgcn_wmma_f32_16x16x32_bf16(false, a_c, false, w_c3, (short)0, acc[3], false, false);

    // --- in-register gates + cell update ---
    // C/D layout: VGPR r -> lanes 0-15: (M=m0+r, N=col0+lane); lanes 16-31: M=m0+8+r.
    const int col   = col0 + row;
    const int mbase = m0 + ((lane >> 4) ? 8 : 0);
    const float bfv = bias[col];
    const float biv = bias[HID + col];
    const float bgv = bias[2 * HID + col];
    const float bov = bias[3 * HID + col];
    const bool last = (t + 1 == T_STEPS);

    #pragma unroll
    for (int r = 0; r < 8; ++r) {
        const int m = mbase + r;
        const float f = fast_sigmoid(acc[0][r] + bfv);
        const float i = fast_sigmoid(acc[1][r] + biv);
        const float g = fast_tanh   (acc[2][r] + bgv);
        const float o = fast_sigmoid(acc[3][r] + bov);
        const float cn = f * cA[m * HID + col] + i * g;
        const float h  = o * fast_tanh(cn);

        cB[m * HID + col] = cn;
        out[(size_t)t * (BATCH * HID) + m * HID + col] = h;            // stacked[t]
        xhB[(size_t)m * KDIM + HID + col] = __float2bfloat16(h);       // h for t+1
        if (!last) {
            xhB[(size_t)m * KDIM + col] =                               // x_{t+1}
                __float2bfloat16(x_all[(size_t)(t + 1) * (BATCH * DIN) + m * DIN + col]);
        } else {
            hx_out[m * HID + col] = h;
            cx_out[m * HID + col] = cn;
        }
    }
}

// ---------------------------------------------------------------------------
// Host launch
// ---------------------------------------------------------------------------
extern "C" void kernel_launch(void* const* d_in, const int* in_sizes, int n_in,
                              void* d_out, int out_size, void* d_ws, size_t ws_size,
                              hipStream_t stream) {
    const float* x   = (const float*)d_in[0];
    const float* Wf  = (const float*)d_in[1];
    const float* bf_ = (const float*)d_in[2];
    const float* Wi  = (const float*)d_in[3];
    const float* bi_ = (const float*)d_in[4];
    const float* Wg  = (const float*)d_in[5];
    const float* bg_ = (const float*)d_in[6];
    const float* Wo  = (const float*)d_in[7];
    const float* bo_ = (const float*)d_in[8];
    float* out = (float*)d_out;

    // Workspace layout
    char* ws = (char*)d_ws;
    __hip_bfloat16* Wt   = (__hip_bfloat16*)ws;                            // 4 MiB
    float*          bias = (float*)(ws + (size_t)NDIM * KDIM * 2);         // 8 KiB
    __hip_bfloat16* xh0  = (__hip_bfloat16*)((char*)bias + NDIM * 4);      // 128 KiB
    __hip_bfloat16* xh1  = xh0 + (size_t)BATCH * KDIM;                     // 128 KiB
    float*          c0   = (float*)(xh1 + (size_t)BATCH * KDIM);           // 128 KiB
    float*          c1   = c0 + (size_t)BATCH * HID;                       // 128 KiB

    float* hx_out = out + (size_t)T_STEPS * BATCH * HID;
    float* cx_out = hx_out + (size_t)BATCH * HID;

    // One-time prep
    {
        int n = NDIM * KDIM;
        lstm_pack_weights<<<(n + 255) / 256, 256, 0, stream>>>(Wf, Wi, Wg, Wo,
                                                               bf_, bi_, bg_, bo_,
                                                               Wt, bias);
        int m = BATCH * HID;
        lstm_init_state<<<(m + 255) / 256, 256, 0, stream>>>(x, xh0, c0);
    }

    __hip_bfloat16* xhbuf[2] = {xh0, xh1};
    float*          cbuf[2]  = {c0, c1};

    // Recurrence: stream order encodes the serial dependency; state ping-pongs.
    for (int t = 0; t < T_STEPS; ++t) {
        const int s = t & 1, d = s ^ 1;
        lstm_step<<<16, 256, 0, stream>>>(xhbuf[s], xhbuf[d], Wt, bias,
                                          cbuf[s], cbuf[d], x, out, t,
                                          hx_out, cx_out);
    }
}
